// MultiHeadedRelAttention_37520834298519
// MI455X (gfx1250) — compile-verified, matrix operand order checked by reading, not run on hardware
//
#include <hip/hip_runtime.h>
#include <hip/hip_bf16.h>

typedef __attribute__((ext_vector_type(16))) __bf16 v16bf;
typedef __attribute__((ext_vector_type(8)))  float  v8f;

#define WMMA_BF16(a, b, c) \
  __builtin_amdgcn_wmma_f32_16x16x32_bf16(false, (a), false, (b), (short)0, (c), false, false)

static constexpr int B_ = 4, L_ = 1024, D_ = 1024, H_ = 16, DH_ = 64, R_ = 32, WST_ = 48;

// LDS byte offset of a generic pointer to __shared__ (low 32 bits = LDS address; ISA 10.2)
#define LDS_OFF(p) ((unsigned)(size_t)(p))

__device__ __forceinline__ void async_g2l_b128(unsigned lds, const void* gaddr) {
  asm volatile("global_load_async_to_lds_b128 %0, %1, off" :: "v"(lds), "v"(gaddr) : "memory");
}
__device__ __forceinline__ void wait_asynccnt0() {
  asm volatile("s_wait_asynccnt 0x0" ::: "memory");
}

// ---------------- fragment loaders (CDNA5 wave32 WMMA layouts) ----------------
__device__ __forceinline__ v16bf ld_a_f32(const float* A, int lda, int lane) {
  int m = lane & 15, ko = (lane >> 4) * 8;
  const float* p = A + (long)m * lda + ko;
  v16bf a;
#pragma unroll
  for (int i = 0; i < 8; ++i) { a[i] = (__bf16)p[i]; a[8 + i] = (__bf16)p[16 + i]; }
  return a;
}
__device__ __forceinline__ v16bf ld_a_bf16(const __bf16* A, int lda, int lane) {
  int m = lane & 15, ko = (lane >> 4) * 8;
  const __bf16* p = A + (long)m * lda + ko;
  v16bf a;
#pragma unroll
  for (int i = 0; i < 8; ++i) { a[i] = p[i]; a[8 + i] = p[16 + i]; }
  return a;
}
// B fragment from transposed-layout source: element (k,n) at Bt[n*ldb + k]
__device__ __forceinline__ v16bf ld_bT_bf16(const __bf16* Bt, int ldb, int lane) {
  int n = lane & 15, ko = (lane >> 4) * 8;
  const __bf16* p = Bt + (long)n * ldb + ko;
  v16bf b;
#pragma unroll
  for (int i = 0; i < 8; ++i) { b[i] = p[i]; b[8 + i] = p[16 + i]; }
  return b;
}

// ---------------- one-time precision/layout prep ----------------
__global__ __launch_bounds__(256) void cvt_bf16(const float* in, __bf16* out) {
  long base = ((long)blockIdx.x * 256 + threadIdx.x) * 4;
  float4 f = *(const float4*)(in + base);
  out[base + 0] = (__bf16)f.x; out[base + 1] = (__bf16)f.y;
  out[base + 2] = (__bf16)f.z; out[base + 3] = (__bf16)f.w;
}

// Wt[n,k] = (bf16)W[k,n]   (64x64 tiles via padded LDS)
__global__ __launch_bounds__(256) void transpose_cvt(const float* W, __bf16* Wt) {
  __shared__ float tile[64][65];
  int bx = blockIdx.x & 15, by = blockIdx.x >> 4;
  int k0 = by * 64, n0 = bx * 64;
  int c = threadIdx.x & 63, r0 = threadIdx.x >> 6;
#pragma unroll
  for (int j = 0; j < 16; ++j) {
    int r = r0 + j * 4;
    tile[r][c] = W[(long)(k0 + r) * D_ + n0 + c];
  }
  __syncthreads();
#pragma unroll
  for (int j = 0; j < 16; ++j) {
    int r = r0 + j * 4;
    Wt[(long)(n0 + r) * D_ + k0 + c] = (__bf16)tile[c][r];
  }
}

// ---------------- kernels 1/5: async-DMA double-buffered bf16 GEMM ----------------
// out = A @ W + bias, A bf16 (B*L x D), W given pre-transposed bf16 Wt[n,k].
// Block tile 128(M) x 64(N), BK=32, 8 waves in 4x2 grid, 2x2 16x16 tiles per wave.
// mode 0: Q -> qu=(val*scale+u), qv=(val*scale+v), bf16 (b,h,l,dh)
// mode 1: K -> bf16 (b,h,l,dh)
// mode 2: V -> bf16 transposed (b,h,dh,l)
// mode 3: O -> fp32 (B*L, D)
__global__ __launch_bounds__(256) void gemm_bf16(
    const __bf16* Ab, const __bf16* Wt, const float* bias,
    const float* uvec, const float* vvec, float scale,
    __bf16* out0, __bf16* out1, float* outF, int mode) {
  __shared__ alignas(16) __bf16 sA[2][128 * 32];
  __shared__ alignas(16) __bf16 sB[2][64 * 32];

  const int tid = threadIdx.x;
  const int lane = tid & 31, wave = tid >> 5;
  const int bm = blockIdx.x >> 4;               // M/128 = 32
  const int bn = blockIdx.x & 15;               // N/64  = 16
  const int wm = wave >> 1, wn = wave & 1;      // 4x2 wave grid
  const int n = lane & 15, ko = (lane >> 4) * 8, hi = lane >> 4;

  // staging map: 16B (8 bf16) chunks
  const int arow = tid >> 2, apart = (tid & 3) * 8;   // A rows 0..63 (+64 for 2nd chunk)
  const int brow = tid >> 2, bpart = (tid & 3) * 8;   // B n-rows 0..63
  const __bf16* Ag = Ab + ((long)bm * 128) * D_;
  const __bf16* Bg = Wt + ((long)bn * 64) * D_;

  auto stage = [&](int kb, int buf) {
    async_g2l_b128(LDS_OFF(&sA[buf][arow * 32 + apart]),
                   Ag + (long)arow * D_ + kb + apart);
    async_g2l_b128(LDS_OFF(&sA[buf][(arow + 64) * 32 + apart]),
                   Ag + (long)(arow + 64) * D_ + kb + apart);
    async_g2l_b128(LDS_OFF(&sB[buf][brow * 32 + bpart]),
                   Bg + (long)brow * D_ + kb + bpart);
  };

  v8f acc[2][2] = {};
  stage(0, 0);
  for (int it = 0; it < 32; ++it) {
    int cur = it & 1;
    wait_asynccnt0();          // own async slice of buf[cur] landed
    __syncthreads();           // all slices landed; all reads of buf[1-cur] done
    if (it < 31) stage((it + 1) * 32, 1 - cur);

    v16bf afr[2], bfr[2];
#pragma unroll
    for (int t = 0; t < 2; ++t) {
      const __bf16* pa = &sA[cur][(wm * 32 + t * 16 + n) * 32 + ko];
      v16bf a;
#pragma unroll
      for (int i = 0; i < 8; ++i) { a[i] = pa[i]; a[8 + i] = pa[16 + i]; }
      afr[t] = a;
      const __bf16* pb = &sB[cur][(wn * 32 + t * 16 + n) * 32 + ko];
      v16bf b;
#pragma unroll
      for (int i = 0; i < 8; ++i) { b[i] = pb[i]; b[8 + i] = pb[16 + i]; }
      bfr[t] = b;
    }
#pragma unroll
    for (int im = 0; im < 2; ++im)
#pragma unroll
      for (int in = 0; in < 2; ++in)
        acc[im][in] = WMMA_BF16(afr[im], bfr[in], acc[im][in]);
  }

  // ---- epilogue ----
  const int row0 = bm * 128 + wm * 32;
  const int col0 = bn * 64 + wn * 32;
#pragma unroll
  for (int im = 0; im < 2; ++im)
#pragma unroll
    for (int in = 0; in < 2; ++in) {
      int colb = col0 + in * 16 + n;
      float bia = bias[colb];
#pragma unroll
      for (int r = 0; r < 8; ++r) {
        int row = row0 + im * 16 + r + 8 * hi;
        float val = acc[im][in][r] + bia;
        if (mode == 3) {
          outF[(long)row * D_ + colb] = val;
        } else {
          int bi = row >> 10, l = row & (L_ - 1);
          int h = colb >> 6, dh = colb & 63;
          if (mode == 0) {
            float q = val * scale;
            long idx = (((long)(bi * H_ + h)) * L_ + l) * DH_ + dh;
            out0[idx] = (__bf16)(q + uvec[colb]);
            out1[idx] = (__bf16)(q + vvec[colb]);
          } else if (mode == 1) {
            long idx = (((long)(bi * H_ + h)) * L_ + l) * DH_ + dh;
            out0[idx] = (__bf16)val;
          } else {  // mode 2: V transposed (b,h,dh,l)
            long idx = (((long)(bi * H_ + h)) * DH_ + dh) * L_ + l;
            out0[idx] = (__bf16)val;
          }
        }
      }
    }
}

// ---------------- kernel 2: w[b,h,l,r] = sum_dh qv * rel[r,h,dh]  (N padded to 48) ----------------
__global__ __launch_bounds__(256) void relw_kernel(const __bf16* qv, const float* rel, float* wbuf) {
  int lane = threadIdx.x & 31;
  int wave = threadIdx.x >> 5;
  int tile = blockIdx.x * 8 + wave;
  int nt = tile % 3;
  int mt = (tile / 3) & 63;
  int bh = tile / 192;
  int h = bh & (H_ - 1);
  int n = lane & 15, ko = (lane >> 4) * 8;
  int r = nt * 16 + n;

  v8f acc = {};
#pragma unroll
  for (int kb = 0; kb < DH_; kb += 32) {
    v16bf a = ld_a_bf16(qv + ((long)bh * L_ + mt * 16) * DH_ + kb, DH_, lane);
    v16bf b;
    if (r <= R_) {
      const float* p = rel + (long)r * D_ + h * DH_ + kb + ko;
#pragma unroll
      for (int i = 0; i < 8; ++i) { b[i] = (__bf16)p[i]; b[8 + i] = (__bf16)p[16 + i]; }
    } else {
#pragma unroll
      for (int i = 0; i < 16; ++i) b[i] = (__bf16)0.0f;
    }
    acc = WMMA_BF16(a, b, acc);
  }
  int hi = lane >> 4;
#pragma unroll
  for (int rr = 0; rr < 8; ++rr) {
    int l = mt * 16 + rr + 8 * hi;
    wbuf[((long)bh * L_ + l) * WST_ + r] = acc[rr];
  }
}

// ---------------- kernel 3: scores + gather + mask + softmax -> attn (d_out) ----------------
__global__ __launch_bounds__(128) void attn_kernel(
    const __bf16* qu, const __bf16* kbuf, const float* wbuf,
    const int* dist, const unsigned char* mask, float* attn_out) {
  int lane = threadIdx.x & 31;
  int wave = threadIdx.x >> 5;
  int qt = blockIdx.x & 63;
  int bh = blockIdx.x >> 6;
  int b = bh >> 4;
  long qbase = (long)bh * L_ + qt * 16;

  v16bf a0 = ld_a_bf16(qu + qbase * DH_ + 0, DH_, lane);
  v16bf a1 = ld_a_bf16(qu + qbase * DH_ + 32, DH_, lane);

  int keybase = wave * 256;
  int n = lane & 15, hi = lane >> 4;
  const int* distb = dist + (long)b * L_ * L_;
  const unsigned char* maskb = mask + (long)b * L_ * L_;
  const float* wb = wbuf + (long)bh * L_ * WST_;

  v8f acc[16];
#pragma unroll
  for (int kt = 0; kt < 16; ++kt) {
    int k0 = keybase + kt * 16;
    v16bf b0 = ld_bT_bf16(kbuf + ((long)bh * L_ + k0) * DH_ + 0, DH_, lane);
    v16bf b1 = ld_bT_bf16(kbuf + ((long)bh * L_ + k0) * DH_ + 32, DH_, lane);
    v8f c = {};
    c = WMMA_BF16(a0, b0, c);
    c = WMMA_BF16(a1, b1, c);
    int key = k0 + n;
#pragma unroll
    for (int r = 0; r < 8; ++r) {
      int q = qt * 16 + r + 8 * hi;
      float s = c[r] + wb[(long)q * WST_ + distb[(long)q * L_ + key]];
      if (maskb[(long)q * L_ + key]) s = -1e18f;
      c[r] = s;
    }
    acc[kt] = c;
  }

  __shared__ float s_max[4][16];
  __shared__ float s_sum[4][16];

  float rm[8];
#pragma unroll
  for (int r = 0; r < 8; ++r) {
    float m = -3.0e38f;
#pragma unroll
    for (int kt = 0; kt < 16; ++kt) m = fmaxf(m, acc[kt][r]);
#pragma unroll
    for (int sh = 1; sh < 16; sh <<= 1) m = fmaxf(m, __shfl_xor(m, sh, 32));
    rm[r] = m;
  }
  if (n == 0) {
#pragma unroll
    for (int r = 0; r < 8; ++r) s_max[wave][hi * 8 + r] = rm[r];
  }
  __syncthreads();
  float gm[8];
#pragma unroll
  for (int r = 0; r < 8; ++r)
    gm[r] = fmaxf(fmaxf(s_max[0][hi * 8 + r], s_max[1][hi * 8 + r]),
                  fmaxf(s_max[2][hi * 8 + r], s_max[3][hi * 8 + r]));

  float ls[8];
#pragma unroll
  for (int r = 0; r < 8; ++r) {
    float s = 0.f;
#pragma unroll
    for (int kt = 0; kt < 16; ++kt) {
      float e = __expf(acc[kt][r] - gm[r]);
      acc[kt][r] = e;
      s += e;
    }
#pragma unroll
    for (int sh = 1; sh < 16; sh <<= 1) s += __shfl_xor(s, sh, 32);
    ls[r] = s;
  }
  if (n == 0) {
#pragma unroll
    for (int r = 0; r < 8; ++r) s_sum[wave][hi * 8 + r] = ls[r];
  }
  __syncthreads();
#pragma unroll
  for (int r = 0; r < 8; ++r) {
    float inv = 1.f / (s_sum[0][hi * 8 + r] + s_sum[1][hi * 8 + r] +
                       s_sum[2][hi * 8 + r] + s_sum[3][hi * 8 + r]);
    int q = qt * 16 + r + 8 * hi;
    float* op = attn_out + ((long)bh * L_ + q) * L_;
#pragma unroll
    for (int kt = 0; kt < 16; ++kt)
      op[keybase + kt * 16 + n] = acc[kt][r] * inv;
  }
}

// ---------------- kernel 4: ctx = attn @ v  (V stored transposed (b,h,dh,l)) ----------------
__global__ __launch_bounds__(128) void ctx_kernel(const float* attn, const __bf16* vtbuf, __bf16* ctx) {
  int lane = threadIdx.x & 31;
  int nt = threadIdx.x >> 5;
  int mt = blockIdx.x & 63;
  int bh = blockIdx.x >> 6;
  int b = bh >> 4, h = bh & (H_ - 1);
  const float* ab = attn + ((long)bh * L_ + mt * 16) * L_;
  const __bf16* vt = vtbuf + ((long)bh * DH_ + nt * 16) * L_;

  v8f acc = {};
#pragma unroll 2
  for (int kb = 0; kb < L_; kb += 32) {
    v16bf a = ld_a_f32(ab + kb, L_, lane);
    v16bf bf = ld_bT_bf16(vt + kb, L_, lane);
    acc = WMMA_BF16(a, bf, acc);
  }
  int n = lane & 15, hi = lane >> 4;
#pragma unroll
  for (int r = 0; r < 8; ++r) {
    int l = mt * 16 + r + 8 * hi;
    int col = h * DH_ + nt * 16 + n;
    ctx[((long)(b * L_ + l)) * D_ + col] = (__bf16)acc[r];
  }
}

// ---------------- host ----------------
extern "C" void kernel_launch(void* const* d_in, const int* in_sizes, int n_in,
                              void* d_out, int out_size, void* d_ws, size_t ws_size,
                              hipStream_t stream) {
  const float* inputs = (const float*)d_in[0];
  const unsigned char* mask = (const unsigned char*)d_in[1];
  const int* distances = (const int*)d_in[2];
  const float* Wq = (const float*)d_in[3];
  const float* bq = (const float*)d_in[4];
  const float* Wk = (const float*)d_in[5];
  const float* bk = (const float*)d_in[6];
  const float* Wv = (const float*)d_in[7];
  const float* bv = (const float*)d_in[8];
  const float* Wo = (const float*)d_in[9];
  const float* bo = (const float*)d_in[10];
  const float* rel = (const float*)d_in[11];
  const float* uvec = (const float*)d_in[12];
  const float* vvec = (const float*)d_in[13];

  float* out = (float*)d_out;
  float* attn_out = out + (long)B_ * L_ * D_;

  char* ws = (char*)d_ws;
  const long nqk = (long)B_ * H_ * L_ * DH_;                 // 4M elems
  const long nw = (long)D_ * D_;                             // 1M elems
  __bf16* xb  = (__bf16*)ws;  ws += nqk * 2;                 // inputs bf16
  __bf16* Wqt = (__bf16*)ws;  ws += nw * 2;
  __bf16* Wkt = (__bf16*)ws;  ws += nw * 2;
  __bf16* Wvt = (__bf16*)ws;  ws += nw * 2;
  __bf16* Wot = (__bf16*)ws;  ws += nw * 2;
  __bf16* qu  = (__bf16*)ws;  ws += nqk * 2;
  __bf16* qv  = (__bf16*)ws;  ws += nqk * 2;
  __bf16* kb  = (__bf16*)ws;  ws += nqk * 2;
  __bf16* vt  = (__bf16*)ws;  ws += nqk * 2;                 // V transposed
  float*  wbuf = (float*)ws;  ws += (long)B_ * H_ * L_ * WST_ * 4;
  __bf16* ctxb = (__bf16*)ws; ws += nqk * 2;

  const float scale = 0.125f;                                // 1/sqrt(64)
  const int gemm_blocks = (B_ * L_ / 128) * (D_ / 64);       // 512

  cvt_bf16<<<(int)(nqk / 1024), 256, 0, stream>>>(inputs, xb);
  transpose_cvt<<<256, 256, 0, stream>>>(Wq, Wqt);
  transpose_cvt<<<256, 256, 0, stream>>>(Wk, Wkt);
  transpose_cvt<<<256, 256, 0, stream>>>(Wv, Wvt);
  transpose_cvt<<<256, 256, 0, stream>>>(Wo, Wot);

  gemm_bf16<<<gemm_blocks, 256, 0, stream>>>(xb, Wqt, bq, uvec, vvec, scale, qu, qv, nullptr, 0);
  gemm_bf16<<<gemm_blocks, 256, 0, stream>>>(xb, Wkt, bk, nullptr, nullptr, 1.f, kb, nullptr, nullptr, 1);
  gemm_bf16<<<gemm_blocks, 256, 0, stream>>>(xb, Wvt, bv, nullptr, nullptr, 1.f, vt, nullptr, nullptr, 2);

  relw_kernel<<<(B_ * H_ * 64 * 3) / 8, 256, 0, stream>>>(qv, rel, wbuf);

  attn_kernel<<<B_ * H_ * 64, 128, 0, stream>>>(qu, kb, wbuf, distances, mask, attn_out);

  ctx_kernel<<<B_ * H_ * 64, 128, 0, stream>>>(attn_out, vt, ctxb);

  gemm_bf16<<<gemm_blocks, 256, 0, stream>>>(ctxb, Wot, bo, nullptr, nullptr, 1.f,
                                             nullptr, nullptr, out, 3);
}
